// LinOSSAttention_68521908240569
// MI455X (gfx1250) — matrix operand order repaired and checked by reading, not compiled, not run on hardware
//
#include <hip/hip_runtime.h>
#include <cstdint>

typedef __bf16 bf16_t;
typedef __bf16 v16bf __attribute__((ext_vector_type(16)));
typedef float  v8f   __attribute__((ext_vector_type(8)));

__device__ __forceinline__ bf16_t f2bf(float f) { return (bf16_t)f; }

// ---------------- pack kernels (fp32 -> bf16, planar weight layouts) ----------------
__global__ void k_cvt_f32_bf16(const float* __restrict__ s, bf16_t* __restrict__ d, int n) {
  int i = blockIdx.x * blockDim.x + threadIdx.x;
  if (i < n) d[i] = f2bf(s[i]);
}

// rows 0..1023 = Wi, rows 1024..2047 = Wg  (2048 x 1024)
__global__ void k_pack_ig(const float* __restrict__ Wi, const float* __restrict__ Wg,
                          bf16_t* __restrict__ d) {
  int i = blockIdx.x * blockDim.x + threadIdx.x;
  int n = i >> 10, k = i & 1023;
  float v = (n < 1024) ? Wi[n * 1024 + k] : Wg[(n - 1024) * 1024 + k];
  d[i] = f2bf(v);
}

// B_param (P,H,2) -> rows 0..1023 = B_re[p,:], rows 1024..2047 = B_im[p,:]
__global__ void k_pack_B(const float* __restrict__ Bp, bf16_t* __restrict__ d) {
  int i = blockIdx.x * blockDim.x + threadIdx.x;
  int n = i >> 10, k = i & 1023;
  int p = n & 1023, c = n >> 10;
  d[i] = f2bf(Bp[((p << 10) + k) * 2 + c]);
}

// C_param (H,P,2) -> (1024 x 2048): cols 0..1023 = C_re[h,:], cols 1024..2047 = -C_im[h,:]
__global__ void k_pack_C(const float* __restrict__ Cp, bf16_t* __restrict__ d) {
  int i = blockIdx.x * blockDim.x + threadIdx.x;
  int h = i >> 11, j = i & 2047;
  int p = j & 1023, c = j >> 10;
  float v = Cp[((h << 10) + p) * 2 + c];
  d[i] = f2bf(c ? -v : v);
}

// ---------------- WMMA GEMM: C(MxN) = A(MxK) * W(NxK)^T ----------------
// 128x128 block tile, 256 threads = 8 waves, each wave a 32x64 sub-tile (8 WMMAs/K-step).
// Double-buffered LDS staged via async global->LDS loads; rows padded to 40 bf16 (80 B).
union FragCast { struct { uint4 a, b; } q; v16bf v; };

// A operand (16x32 bf16): lane L<16 -> M=L, K {0..7,16..23}; L>=16 -> +8 on K
__device__ __forceinline__ v16bf loadA_frag(const bf16_t* base, int rl, int hf) {
  const char* p = (const char*)base + rl * 80 + hf * 16;
  FragCast u;
  u.q.a = *(const uint4*)(p);
  u.q.b = *(const uint4*)(p + 32);
  return u.v;
}
// B operand (32x16 bf16): lane L -> N=L&15, K {0..7,16..23}; L>=16 -> +8 on K
__device__ __forceinline__ v16bf loadB_frag(const bf16_t* base, int rl, int hf) {
  const char* p = (const char*)base + rl * 80 + hf * 16;
  FragCast u;
  u.q.a = *(const uint4*)(p);
  u.q.b = *(const uint4*)(p + 32);
  return u.v;
}

// CDNA5 async global->LDS staging (ASYNCcnt tracked; loads complete in order)
__device__ __forceinline__ void async_ld_b128(unsigned lds_addr, const bf16_t* gptr) {
  asm volatile("global_load_async_to_lds_b128 %0, %1, off" ::"v"(lds_addr), "v"(gptr)
               : "memory");
}
__device__ __forceinline__ void wait_async4() {  // oldest tile (4 loads/thread) landed
  asm volatile("s_wait_asynccnt 0x4" ::: "memory");
}
__device__ __forceinline__ void wait_async0() {
  asm volatile("s_wait_asynccnt 0x0" ::: "memory");
}

// EPI: 0 = store bf16, 1 = store f32, 2 = store f32 with o += D[n]*x[m,n]
template <int EPI>
__global__ __launch_bounds__(256) void k_gemm_wmma(
    const bf16_t* __restrict__ A, int lda, const bf16_t* __restrict__ W, int ldw,
    void* __restrict__ Cout, int ldc, int K, const float* __restrict__ Dvec,
    const bf16_t* __restrict__ Xmat, int ldx) {
  __shared__ bf16_t As[2][128 * 40];
  __shared__ bf16_t Ws[2][128 * 40];

  const int tid = threadIdx.x;
  const int lane = tid & 31, wave = tid >> 5;
  const int wm = (wave >> 1) << 5;  // 0,32,64,96
  const int wn = (wave & 1) << 6;   // 0,64
  const int rl = lane & 15, hf = lane >> 4;

  const int rowM = blockIdx.y << 7;
  const int rowN = blockIdx.x << 7;

  v8f acc[2][4] = {};

  // stage one 128x32 A tile + 128x32 W tile: 4 async b128 loads per thread
  auto stage = [&](bf16_t* Asb, bf16_t* Wsb, int kk2, bool pf) {
#pragma unroll
    for (int t = 0; t < 2; ++t) {
      int q = tid + (t << 8);        // 0..511
      int row = q >> 2, ch = q & 3;  // 4x b128 chunks per 32-wide row
      const bf16_t* ga = A + (size_t)(rowM + row) * lda + kk2 + (ch << 3);
      const bf16_t* gw = W + (size_t)(rowN + row) * ldw + kk2 + (ch << 3);
      if (pf && ch == 0) {  // prefetch one further K tile (first-touch HBM latency)
        __builtin_prefetch((const char*)ga + 64, 0, 1);
        __builtin_prefetch((const char*)gw + 64, 0, 1);
      }
      async_ld_b128((unsigned)(uintptr_t)&Asb[row * 40 + (ch << 3)], ga);
      async_ld_b128((unsigned)(uintptr_t)&Wsb[row * 40 + (ch << 3)], gw);
    }
  };

  auto compute = [&](const bf16_t* Asb, const bf16_t* Wsb) {
    v16bf a0 = loadA_frag(Asb + (wm + 0) * 40, rl, hf);
    v16bf a1 = loadA_frag(Asb + (wm + 16) * 40, rl, hf);
    v16bf b0 = loadB_frag(Wsb + (wn + 0) * 40, rl, hf);
    v16bf b1 = loadB_frag(Wsb + (wn + 16) * 40, rl, hf);
    v16bf b2 = loadB_frag(Wsb + (wn + 32) * 40, rl, hf);
    v16bf b3 = loadB_frag(Wsb + (wn + 48) * 40, rl, hf);
    acc[0][0] = __builtin_amdgcn_wmma_f32_16x16x32_bf16(false, a0, false, b0, (short)0, acc[0][0], false, false);
    acc[0][1] = __builtin_amdgcn_wmma_f32_16x16x32_bf16(false, a0, false, b1, (short)0, acc[0][1], false, false);
    acc[0][2] = __builtin_amdgcn_wmma_f32_16x16x32_bf16(false, a0, false, b2, (short)0, acc[0][2], false, false);
    acc[0][3] = __builtin_amdgcn_wmma_f32_16x16x32_bf16(false, a0, false, b3, (short)0, acc[0][3], false, false);
    acc[1][0] = __builtin_amdgcn_wmma_f32_16x16x32_bf16(false, a1, false, b0, (short)0, acc[1][0], false, false);
    acc[1][1] = __builtin_amdgcn_wmma_f32_16x16x32_bf16(false, a1, false, b1, (short)0, acc[1][1], false, false);
    acc[1][2] = __builtin_amdgcn_wmma_f32_16x16x32_bf16(false, a1, false, b2, (short)0, acc[1][2], false, false);
    acc[1][3] = __builtin_amdgcn_wmma_f32_16x16x32_bf16(false, a1, false, b3, (short)0, acc[1][3], false, false);
  };

  // double-buffered pipeline; K is a multiple of 64
  stage(As[0], Ws[0], 0, true);
  for (int kk = 0; kk < K; kk += 64) {
    stage(As[1], Ws[1], kk + 32, kk + 64 < K);
    wait_async4();  // in-order completion: buffer0's 4 loads are done
    __syncthreads();
    compute(As[0], Ws[0]);
    __syncthreads();
    if (kk + 64 < K) {
      stage(As[0], Ws[0], kk + 64, kk + 96 < K);
      wait_async4();  // buffer1's loads are done
    } else {
      wait_async0();
    }
    __syncthreads();
    compute(As[1], Ws[1]);
    __syncthreads();
  }

  // epilogue: C/D layout -> lane<16: (M=r, N=lane); lane>=16: (M=8+r, N=lane-16)
#pragma unroll
  for (int i = 0; i < 2; ++i) {
#pragma unroll
    for (int j = 0; j < 4; ++j) {
      int mm0 = rowM + wm + i * 16 + (hf << 3);
      int nn = rowN + wn + j * 16 + rl;
#pragma unroll
      for (int r = 0; r < 8; ++r) {
        int mm = mm0 + r;
        float val = acc[i][j][r];
        if (EPI == 2) val += Dvec[nn] * (float)Xmat[(size_t)mm * ldx + nn];
        if (EPI == 0)
          ((bf16_t*)Cout)[(size_t)mm * ldc + nn] = f2bf(val);
        else
          ((float*)Cout)[(size_t)mm * ldc + nn] = val;
      }
    }
  }
}

// ---------------- LinOSS scan (IM discretization), 3-phase chunked ----------------
// state s = (zr, zi, yr, yi); s_{t+1} = M s_t + F bu_t, M/F constant per channel p.
#define SCAN_T 2048
#define SCAN_P 1024
#define SCAN_NC 16   // chunks
#define SCAN_TC 128  // T per chunk (2^7)

__device__ __forceinline__ void scan_coefs(const float* Ad, const float* dtp, int p,
                                           float& m11, float& m12, float& m21, float& m22,
                                           float& f1, float& f2) {
  float Aa = fmaxf(Ad[p], 0.0f);
  float dt = 1.0f / (1.0f + __expf(-dtp[p]));
  float S = 1.0f / (1.0f + dt * dt * Aa);
  m11 = S; m12 = -dt * Aa * S; m21 = dt * S; m22 = S;
  f1 = dt * S; f2 = dt * dt * S;
}

// phase 1: per (b,chunk,p) run local recurrence from zero, emit partial end-state
__global__ void k_scan_part(const float* __restrict__ Bu, float* __restrict__ part,
                            const float* __restrict__ Ad, const float* __restrict__ dtp) {
  int idx = blockIdx.x * blockDim.x + threadIdx.x;  // B*NC*P
  int p = idx & (SCAN_P - 1);
  int r = idx >> 10;
  int c = r & (SCAN_NC - 1);
  int b = r >> 4;
  float m11, m12, m21, m22, f1, f2;
  scan_coefs(Ad, dtp, p, m11, m12, m21, m22, f1, f2);
  float zr = 0.f, zi = 0.f, yr = 0.f, yi = 0.f;
  const float* bu = Bu + ((size_t)b * SCAN_T + c * SCAN_TC) * 2048 + p;
  for (int t = 0; t < SCAN_TC; ++t) {
    float br = bu[0], bi = bu[1024];
    float z1r = m11 * zr + m12 * yr + f1 * br;
    float z1i = m11 * zi + m12 * yi + f1 * bi;
    float y1r = m21 * zr + m22 * yr + f2 * br;
    float y1i = m21 * zi + m22 * yi + f2 * bi;
    zr = z1r; zi = z1i; yr = y1r; yi = y1i;
    bu += 2048;
  }
  *(float4*)&part[(size_t)idx * 4] = make_float4(zr, zi, yr, yi);
}

// phase 2: per (b,p) chain 16 chunk transitions using Mpow = M^TC (7 squarings)
__global__ void k_scan_combine(const float* __restrict__ part, float* __restrict__ init,
                               const float* __restrict__ Ad, const float* __restrict__ dtp) {
  int idx = blockIdx.x * blockDim.x + threadIdx.x;  // B*P
  int p = idx & (SCAN_P - 1);
  int b = idx >> 10;
  float m11, m12, m21, m22, f1, f2;
  scan_coefs(Ad, dtp, p, m11, m12, m21, m22, f1, f2);
  float a = m11, bb = m12, cc = m21, d = m22;
#pragma unroll
  for (int s = 0; s < 7; ++s) {  // (2x2)^(2^7) = M^128
    float a2 = a * a + bb * cc;
    float b2 = bb * (a + d);
    float c2 = cc * (a + d);
    float d2 = cc * bb + d * d;
    a = a2; bb = b2; cc = c2; d = d2;
  }
  float zr = 0.f, zi = 0.f, yr = 0.f, yi = 0.f;
  for (int c = 0; c < SCAN_NC; ++c) {
    size_t o = (((size_t)b * SCAN_NC + c) * SCAN_P + p) * 4;
    *(float4*)&init[o] = make_float4(zr, zi, yr, yi);
    float4 pt = *(const float4*)&part[o];
    float z1r = a * zr + bb * yr + pt.x;
    float z1i = a * zi + bb * yi + pt.y;
    float y1r = cc * zr + d * yr + pt.z;
    float y1i = cc * zi + d * yi + pt.w;
    zr = z1r; zi = z1i; yr = y1r; yi = y1i;
  }
}

// phase 3: re-run each chunk from its true initial state, write ys (bf16 planar)
__global__ void k_scan_emit(const float* __restrict__ Bu, const float* __restrict__ init,
                            bf16_t* __restrict__ ys, const float* __restrict__ Ad,
                            const float* __restrict__ dtp) {
  int idx = blockIdx.x * blockDim.x + threadIdx.x;  // B*NC*P
  int p = idx & (SCAN_P - 1);
  int r = idx >> 10;
  int c = r & (SCAN_NC - 1);
  int b = r >> 4;
  float m11, m12, m21, m22, f1, f2;
  scan_coefs(Ad, dtp, p, m11, m12, m21, m22, f1, f2);
  float4 s0 = *(const float4*)&init[(size_t)idx * 4];
  float zr = s0.x, zi = s0.y, yr = s0.z, yi = s0.w;
  const float* bu = Bu + ((size_t)b * SCAN_T + c * SCAN_TC) * 2048 + p;
  bf16_t* yo = ys + ((size_t)b * SCAN_T + c * SCAN_TC) * 2048 + p;
  for (int t = 0; t < SCAN_TC; ++t) {
    float br = bu[0], bi = bu[1024];
    float z1r = m11 * zr + m12 * yr + f1 * br;
    float z1i = m11 * zi + m12 * yi + f1 * bi;
    float y1r = m21 * zr + m22 * yr + f2 * br;
    float y1i = m21 * zi + m22 * yi + f2 * bi;
    zr = z1r; zi = z1i; yr = y1r; yi = y1i;
    yo[0] = f2bf(yr);
    yo[1024] = f2bf(yi);
    bu += 2048;
    yo += 2048;
  }
}

// ---------------- gated RMSNorm: v = rmsnorm(o)*w * g*sigmoid(g) ----------------
__global__ __launch_bounds__(256) void k_norm(const float* __restrict__ o,    // (M,1024)
                                              const bf16_t* __restrict__ xg,  // (M,2048)
                                              const float* __restrict__ nw,
                                              bf16_t* __restrict__ v) {
  int m = blockIdx.x;
  const float* orow = o + (size_t)m * 1024;
  float vals[4], ss = 0.f;
#pragma unroll
  for (int i = 0; i < 4; ++i) {
    float f = orow[threadIdx.x + i * 256];
    vals[i] = f;
    ss += f * f;
  }
  __shared__ float red[256];
  red[threadIdx.x] = ss;
  __syncthreads();
  for (int s = 128; s > 0; s >>= 1) {
    if (threadIdx.x < s) red[threadIdx.x] += red[threadIdx.x + s];
    __syncthreads();
  }
  float rstd = rsqrtf(red[0] * (1.0f / 1024.0f) + 1e-5f);
  const bf16_t* grow = xg + (size_t)m * 2048 + 1024;
  bf16_t* vrow = v + (size_t)m * 1024;
#pragma unroll
  for (int i = 0; i < 4; ++i) {
    int h = threadIdx.x + i * 256;
    float g = (float)grow[h];
    float sw = g / (1.0f + __expf(-g));
    vrow[h] = f2bf(vals[i] * rstd * nw[h] * sw);
  }
}

// ---------------- launcher ----------------
extern "C" void kernel_launch(void* const* d_in, const int* in_sizes, int n_in,
                              void* d_out, int out_size, void* d_ws, size_t ws_size,
                              hipStream_t stream) {
  const float* hs = (const float*)d_in[0];
  const float* Wi = (const float*)d_in[1];
  const float* Wg = (const float*)d_in[2];
  const float* Wo = (const float*)d_in[3];
  const float* Ad = (const float*)d_in[4];
  const float* Bp = (const float*)d_in[5];
  const float* Cp = (const float*)d_in[6];
  const float* Dv = (const float*)d_in[7];
  const float* dtp = (const float*)d_in[8];
  const float* nw = (const float*)d_in[9];

  constexpr int Bsz = 4, T = 2048, HID = 1024, H = 1024, P = 1024;
  constexpr size_t M = (size_t)Bsz * T;  // 8192 rows

  char* w = (char*)d_ws;
  auto take = [&](size_t bytes) {
    char* p = w;
    w += (bytes + 255) & ~(size_t)255;
    return p;
  };
  bf16_t* hs_bf = (bf16_t*)take(M * HID * 2);               // 16 MB
  bf16_t* W_ig  = (bf16_t*)take((size_t)2 * H * HID * 2);   // 4 MB
  bf16_t* W_B   = (bf16_t*)take((size_t)2 * P * H * 2);     // 4 MB
  bf16_t* W_C   = (bf16_t*)take((size_t)H * 2 * P * 2);     // 4 MB
  bf16_t* W_o   = (bf16_t*)take((size_t)HID * H * 2);       // 2 MB
  bf16_t* xg    = (bf16_t*)take(M * 2048 * 2);              // 32 MB
  float*  Bu    = (float*)take(M * 2048 * 4);               // 64 MB
  bf16_t* ysb   = (bf16_t*)take(M * 2048 * 2);              // 32 MB
  bf16_t* vb    = (bf16_t*)take(M * 1024 * 2);              // 16 MB
  float*  part  = (float*)take((size_t)Bsz * SCAN_NC * P * 4 * 4);  // 1 MB
  float*  sini  = (float*)take((size_t)Bsz * SCAN_NC * P * 4 * 4);  // 1 MB
  float*  o_f   = Bu;  // alias: Bu dead after the scan

  const int TPB = 256;
  k_cvt_f32_bf16<<<(int)(M * HID / TPB), TPB, 0, stream>>>(hs, hs_bf, (int)(M * HID));
  k_pack_ig<<<2048 * 1024 / TPB, TPB, 0, stream>>>(Wi, Wg, W_ig);
  k_pack_B<<<2048 * 1024 / TPB, TPB, 0, stream>>>(Bp, W_B);
  k_pack_C<<<1024 * 2048 / TPB, TPB, 0, stream>>>(Cp, W_C);
  k_cvt_f32_bf16<<<1024 * 1024 / TPB, TPB, 0, stream>>>(Wo, W_o, 1024 * 1024);

  dim3 blk(256);
  // 1) [x|g] = hs @ [Wi;Wg]^T  -> bf16
  k_gemm_wmma<0><<<dim3(2048 / 128, (int)(M / 128)), blk, 0, stream>>>(
      hs_bf, HID, W_ig, HID, (void*)xg, 2048, HID, nullptr, nullptr, 0);
  // 2) Bu = x @ [B_re;B_im]^T -> f32
  k_gemm_wmma<1><<<dim3(2048 / 128, (int)(M / 128)), blk, 0, stream>>>(
      xg, 2048, W_B, H, (void*)Bu, 2048, H, nullptr, nullptr, 0);
  // 3) chunked LinOSS scan
  k_scan_part<<<Bsz * SCAN_NC * P / TPB, TPB, 0, stream>>>(Bu, part, Ad, dtp);
  k_scan_combine<<<Bsz * P / TPB, TPB, 0, stream>>>(part, sini, Ad, dtp);
  k_scan_emit<<<Bsz * SCAN_NC * P / TPB, TPB, 0, stream>>>(Bu, sini, ysb, Ad, dtp);
  // 4) o = ys @ [C_re|-C_im]^T + D*x -> f32 (aliases Bu)
  k_gemm_wmma<2><<<dim3(1024 / 128, (int)(M / 128)), blk, 0, stream>>>(
      ysb, 2048, W_C, 2048, (void*)o_f, 1024, 2048, Dv, xg, 2048);
  // 5) gated RMSNorm -> bf16
  k_norm<<<(int)M, TPB, 0, stream>>>(o_f, xg, nw, vb);
  // 6) out = v @ Wo^T -> f32
  k_gemm_wmma<1><<<dim3(1024 / 128, (int)(M / 128)), blk, 0, stream>>>(
      vb, 1024, W_o, HID, d_out, 1024, H, nullptr, nullptr, 0);
}
